// HMM_10642928959896
// MI455X (gfx1250) — matrix operand order repaired and checked by reading, not compiled, not run on hardware
//
#include <hip/hip_runtime.h>
#include <hip/hip_bf16.h>
#include <math.h>

typedef __attribute__((ext_vector_type(16))) _Float16 v16h;
typedef __attribute__((ext_vector_type(2)))  _Float16 h2;
typedef __attribute__((ext_vector_type(8)))  float    v8f;

#define BATCH 4096
#define TLEN  1024
#define NS    64

// D = A*B + C  (f16 inputs, f32 accumulate), 16x16x32
#define WMMA_F16(Afr_, B_, C_) \
  __builtin_amdgcn_wmma_f32_16x16x32_f16(false, (Afr_), false, (B_), (short)0, (C_), false, false)

// State relabeling: D-position M=16g+8H+r holds logical state s' with bits
// [5]=g1 [4]=H [3]=g0 [2:0]=r  (swap bits 3 and 4 of M). With this, the D
// fragments map onto next step's B operand with NO cross-lane movement.
__device__ __forceinline__ int perm_m(int m) {
  return (m & 0x07) | ((m & 0x08) << 1) | ((m & 0x10) >> 1) | (m & 0x20);
}

// xor-16 half swap: VALU v_permlanex16 if available, else DS bpermute path.
__device__ __forceinline__ float xhalf_swap(float v) {
#if __has_builtin(__builtin_amdgcn_permlanex16)
  int i = __builtin_bit_cast(int, v);
  i = __builtin_amdgcn_permlanex16(i, i, 0x76543210, 0xfedcba98, false, false);
  return __builtin_bit_cast(float, i);
#else
  return __shfl_xor(v, 16);
#endif
}

__global__ __launch_bounds__(256, 1)
void hmm_forward_kernel(const int*   __restrict__ y,
                        const float* __restrict__ tr,
                        const float* __restrict__ em,
                        const float* __restrict__ start,
                        float*       __restrict__ ws)
{
  __shared__ float Tp[NS * NS];     // row-softmax of transition (linear probs)
  __shared__ float rowlse[NS];
  __shared__ float lEsh[NS * 2];    // log emission probs
  __shared__ float Elin[NS * 2];    // linear emission probs
  __shared__ float lpi[NS];         // log start probs
  __shared__ float slse;
  __shared__ float partial[8];

  const int tid = threadIdx.x;

  // ---------------- block-cooperative parameter normalization ----------------
  if (tid < NS) {
    float mx = -1e30f;
    for (int j = 0; j < NS; ++j) mx = fmaxf(mx, tr[tid * NS + j]);
    float sm = 0.f;
    for (int j = 0; j < NS; ++j) sm += expf(tr[tid * NS + j] - mx);
    rowlse[tid] = mx + logf(sm);

    float e0 = em[tid * 2 + 0], e1 = em[tid * 2 + 1];
    float mm = fmaxf(e0, e1);
    float l  = mm + logf(expf(e0 - mm) + expf(e1 - mm));
    lEsh[tid * 2 + 0] = e0 - l;
    lEsh[tid * 2 + 1] = e1 - l;
    Elin[tid * 2 + 0] = expf(e0 - l);
    Elin[tid * 2 + 1] = expf(e1 - l);
  }
  if (tid == 0) {
    float mx = -1e30f;
    for (int s = 0; s < NS; ++s) mx = fmaxf(mx, start[s]);
    float sm = 0.f;
    for (int s = 0; s < NS; ++s) sm += expf(start[s] - mx);
    slse = mx + logf(sm);
  }
  __syncthreads();
  for (int idx = tid; idx < NS * NS; idx += 256)
    Tp[idx] = expf(tr[idx] - rowlse[idx >> 6]);
  if (tid < NS) lpi[tid] = start[tid] - slse;
  __syncthreads();

  // ---------------- per-lane constant fragments ----------------
  const int  lane = tid & 31;
  const bool lo   = lane < 16;
  const int  bl   = lane & 15;

  // A = P^T with permuted row labels, in A-layout (frag g = M-tile, c = K-chunk)
  v16h Afr[4][2];
#pragma unroll
  for (int g = 0; g < 4; ++g)
#pragma unroll
    for (int c = 0; c < 2; ++c)
#pragma unroll
      for (int j = 0; j < 16; ++j) {
        int kl    = j + (lo ? 0 : 16);                    // K within chunk (A layout)
        int sprev = 32 * c + kl;                          // logical source state
        int sp    = perm_m(16 * g + bl);                  // logical dest state
        Afr[g][c][j] = (_Float16)Tp[sprev * NS + sp];
      }

  // Emission tables, packed f16 pairs in B-half layout:
  // pair p of chunk c covers states s0 = 32c + 16H + 2p, s0+1   (H = lane half)
  int em0i[2][8], em1i[2][8];
#pragma unroll
  for (int c = 0; c < 2; ++c)
#pragma unroll
    for (int p = 0; p < 8; ++p) {
      int s0 = 32 * c + (lo ? 0 : 16) + 2 * p;
      h2 e0; e0.x = (_Float16)Elin[s0 * 2 + 0]; e0.y = (_Float16)Elin[(s0 + 1) * 2 + 0];
      h2 e1; e1.x = (_Float16)Elin[s0 * 2 + 1]; e1.y = (_Float16)Elin[(s0 + 1) * 2 + 1];
      em0i[c][p] = __builtin_bit_cast(int, e0);
      em1i[c][p] = __builtin_bit_cast(int, e1);
    }

  // ---------------- init: a_0 = exp(alpha0 - m0) in B layout ----------------
  const int  b  = blockIdx.x * 128 + (tid >> 5) * 16 + bl;
  const int* yb = y + (size_t)b * TLEN;
  const int  y0 = yb[0];

  float a0[2][16];
  float m0 = -1e30f;
#pragma unroll
  for (int c = 0; c < 2; ++c)
#pragma unroll
    for (int h = 0; h < 16; ++h) {
      int s = 32 * c + (lo ? 0 : 16) + h;
      float v = lpi[s] + lEsh[s * 2 + y0];
      a0[c][h] = v;
      m0 = fmaxf(m0, v);
    }
  m0 = fmaxf(m0, xhalf_swap(m0));

  v16h Bv[2];
#pragma unroll
  for (int c = 0; c < 2; ++c)
#pragma unroll
    for (int h = 0; h < 16; ++h)
      Bv[c][h] = (_Float16)expf(a0[c][h] - m0);

  // ---------------- main scan: 1023 steps of WMMA + rescale ----------------
  int exsum = 0;              // running power-of-2 offset (integer, per batch)
  int ynext = yb[1];          // software-pipelined observation load
  v8f acc[4];
#pragma unroll 1
  for (int t = 1; t < TLEN; ++t) {
    const int yt = ynext;
    int tn = t + 1;
    tn = (tn < TLEN) ? tn : (TLEN - 1);
    ynext = yb[tn];           // consumed next iteration: latency fully hidden

    v8f z = {};
#pragma unroll
    for (int g = 0; g < 4; ++g) {
      v8f p0 = WMMA_F16(Afr[g][0], Bv[0], z);
      acc[g] = WMMA_F16(Afr[g][1], Bv[1], p0);
    }

    // per-batch max over 64 states: balanced tree (short dep chain) + half swap
    float fm[4];
#pragma unroll
    for (int g = 0; g < 4; ++g) {
      float x0 = fmaxf(acc[g][0], acc[g][1]);
      float x1 = fmaxf(acc[g][2], acc[g][3]);
      float x2 = fmaxf(acc[g][4], acc[g][5]);
      float x3 = fmaxf(acc[g][6], acc[g][7]);
      fm[g] = fmaxf(fmaxf(x0, x1), fmaxf(x2, x3));
    }
    float m = fmaxf(fmaxf(fm[0], fm[1]), fmaxf(fm[2], fm[3]));
    m = fmaxf(m, xhalf_swap(m));

    int ex;
    (void)frexpf(m, &ex);                 // m = f * 2^ex, f in [0.5,1)
    float scale = ldexpf(1.0f, -ex);      // power-of-2 rescale, exact
    exsum += ex;

    // rebuild B operand: thanks to the state relabeling, D frag (2c, 2c+1)
    // maps directly onto B chunk c halves (0-7, 8-15) on the SAME lane.
#pragma unroll
    for (int c = 0; c < 2; ++c) {
      v16h nb;
#pragma unroll
      for (int p = 0; p < 8; ++p) {
        int fr = 2 * c + (p >> 2);
        int r0 = (p & 3) * 2;
        h2 q;
        q.x = (_Float16)(acc[fr][r0 + 0] * scale);  // v_fma_mix -> f16
        q.y = (_Float16)(acc[fr][r0 + 1] * scale);
        int ei = yt ? em1i[c][p] : em0i[c][p];      // one cndmask_b32 per pair
        q = q * __builtin_bit_cast(h2, ei);         // v_pk_mul_f16
        nb[2 * p + 0] = q.x;
        nb[2 * p + 1] = q.y;
      }
      Bv[c] = nb;
    }
  }

  // ---------------- finish: log(sum a_T) + c per batch, then reduce ----------------
  float cvec = m0 + (float)exsum * 0.69314718055994531f;

  float ssum = 0.f;
#pragma unroll
  for (int c = 0; c < 2; ++c)
#pragma unroll
    for (int h = 0; h < 16; ++h) ssum += (float)Bv[c][h];
  ssum += xhalf_swap(ssum);              // combine both half-lane state sets
  float lp = logf(ssum) + cvec;          // identical on lane pair (l, l+16)

  // wave sum (each batch counted twice -> *0.5), then block sum
  float wsum = lp;
#pragma unroll
  for (int off = 1; off < 32; off <<= 1) wsum += __shfl_xor(wsum, off);
  wsum *= 0.5f;
  if ((tid & 31) == 0) partial[tid >> 5] = wsum;
  __syncthreads();
  if (tid == 0) {
    float bs = 0.f;
    for (int w = 0; w < 8; ++w) bs += partial[w];
    ws[blockIdx.x] = bs;
  }
}

__global__ void hmm_finalize(const float* __restrict__ ws, float* __restrict__ out) {
  if (threadIdx.x == 0 && blockIdx.x == 0) {
    float s = 0.f;
    for (int i = 0; i < 32; ++i) s += ws[i];
    out[0] = s * (1.0f / (float)BATCH);
  }
}

extern "C" void kernel_launch(void* const* d_in, const int* in_sizes, int n_in,
                              void* d_out, int out_size, void* d_ws, size_t ws_size,
                              hipStream_t stream) {
  (void)in_sizes; (void)n_in; (void)out_size; (void)ws_size;
  const int*   y  = (const int*)d_in[0];
  const float* tr = (const float*)d_in[1];
  const float* em = (const float*)d_in[2];
  const float* st = (const float*)d_in[3];
  float* ws = (float*)d_ws;

  hmm_forward_kernel<<<32, 256, 0, stream>>>(y, tr, em, st, ws);
  hmm_finalize<<<1, 32, 0, stream>>>(ws, (float*)d_out);
}
